// GRUNet_29703993819947
// MI455X (gfx1250) — matrix operand order repaired and checked by reading, not compiled, not run on hardware
//
#include <hip/hip_runtime.h>
#include <hip/hip_bf16.h>

// ---------------------------------------------------------------------------
// GRUNet for MI455X (gfx1250): bf16 WMMA GEMMs + LDS-resident recurrent scan.
// Round 2: strip-grouped scan (4x less W_hh traffic on the serial path),
//          fast TRANS-op gate math, near-scope prefetch.
// ---------------------------------------------------------------------------

typedef __bf16 bf16_t;
typedef __attribute__((ext_vector_type(16))) __bf16 v16bf;
typedef __attribute__((ext_vector_type(8)))  __bf16 v8bf;
typedef __attribute__((ext_vector_type(8)))  float  v8f;

constexpr int kB  = 128;   // batch
constexpr int kS  = 512;   // sequence
constexpr int kE  = 256;   // embed dim
constexpr int kH  = 256;   // hidden
constexpr int kG  = 768;   // 3H (gate dim)
constexpr int kC  = 345;   // classes
constexpr int H_LD = kH + 8;                                   // padded LDS ld
constexpr int SCAN_SMEM = kB * H_LD * 2 + 3 * kB * kH * 2;     // 264,192 B

// --------------------------- WMMA fragment loads ---------------------------

__device__ __forceinline__ v16bf make_frag(const bf16_t* p0, const bf16_t* p1) {
  union { v16bf v; v8bf h[2]; } u;
  u.h[0] = *(const v8bf*)p0;
  u.h[1] = *(const v8bf*)p1;
  return u.v;
}

// A: 16x32 bf16, row-major, leading dim lda.
// lane<16: row=mbase+lane, K = kbase+[0..7] ++ kbase+[16..23]
// lane>=16: row=mbase+lane-16, K = kbase+[8..15] ++ kbase+[24..31]
__device__ __forceinline__ v16bf load_frag_a(const bf16_t* A, int lda,
                                             int mbase, int kbase) {
  const int lane = threadIdx.x & 31;
  const bf16_t* p = A + (size_t)(mbase + (lane & 15)) * lda
                      + kbase + ((lane >> 4) << 3);
  return make_frag(p, p + 16);
}

// A with row remap: actual A row = (mbase + lane15) * rs + ro  (strided M)
__device__ __forceinline__ v16bf load_frag_a_rm(const bf16_t* A, int lda,
                                                int mbase, int kbase,
                                                int rs, long ro) {
  const int lane = threadIdx.x & 31;
  const size_t row = (size_t)(mbase + (lane & 15)) * rs + (size_t)ro;
  const bf16_t* p = A + row * lda + kbase + ((lane >> 4) << 3);
  return make_frag(p, p + 16);
}

// B: 32x16 bf16 taken from W[N][K] row-major (so B = W^T).
// lane<16: col=nbase+lane, K = kbase+[0..7] ++ kbase+[16..23]; lane>=16: K = kbase+[8..15] ++ kbase+[24..31]
__device__ __forceinline__ v16bf load_frag_b(const bf16_t* W, int ldk,
                                             int nbase, int kbase) {
  const int lane = threadIdx.x & 31;
  const bf16_t* p = W + (size_t)(nbase + (lane & 15)) * ldk
                      + kbase + ((lane >> 4) << 3);
  return make_frag(p, p + 16);
}

__device__ __forceinline__ v8f wmma_bf16(v16bf a, v16bf b, v8f c) {
  return __builtin_amdgcn_wmma_f32_16x16x32_bf16(false, a, false, b,
                                                 (short)0, c, false, false);
}

// fast gate math: v_exp_f32 + v_rcp_f32 (TRANS ops, co-execute with XDL)
__device__ __forceinline__ float fast_sigmoid(float x) {
  return __builtin_amdgcn_rcpf(1.0f + __expf(-x));
}
__device__ __forceinline__ float fast_tanh(float x) {
  return 1.0f - 2.0f * __builtin_amdgcn_rcpf(__expf(2.0f * x) + 1.0f);
}

// ------------------------------- small kernels -----------------------------

__global__ void convert_f32_bf16(const float* __restrict__ in,
                                 bf16_t* __restrict__ out, int n) {
  int i = blockIdx.x * blockDim.x + threadIdx.x;
  const int stride = gridDim.x * blockDim.x;
  for (; i < n; i += stride) out[i] = (bf16_t)in[i];
}

// x[b,s,e] = coord·coord_W + feat_emb[flag] + feat_emb[pos]   (bf16 out)
__global__ void embed_kernel(const float* __restrict__ coord,
                             const int* __restrict__ flags,
                             const int* __restrict__ pos,
                             const float* __restrict__ coordW,
                             const float* __restrict__ femb,
                             bf16_t* __restrict__ xout) {
  const int row = blockIdx.x;          // b*S + s  (65536 rows)
  const int e   = threadIdx.x;         // 0..255
  const float c0 = coord[row * 2 + 0];
  const float c1 = coord[row * 2 + 1];
  const int f = flags[row];
  const int p = pos[row];
  const float v = c0 * coordW[e * 2 + 0] + c1 * coordW[e * 2 + 1]
                + femb[f * kE + e] + femb[p * kE + e];
  xout[(size_t)row * kE + e] = (bf16_t)v;
}

// logits[b,c] = featur[b,:] · out_W[c,:] + out_b[c]   (fp32, tiny GEMM)
__global__ void logits_kernel(const float* __restrict__ featur,
                              const float* __restrict__ outW,
                              const float* __restrict__ outb,
                              float* __restrict__ logits) {
  const int idx = blockIdx.x * blockDim.x + threadIdx.x;
  if (idx >= kB * kC) return;
  const int b = idx / kC;
  const int c = idx % kC;
  const float* f = featur + b * (2 * kH);
  const float* w = outW + (size_t)c * (2 * kH);
  float s = outb[c];
  #pragma unroll 8
  for (int k = 0; k < 2 * kH; ++k) s = fmaf(f[k], w[k], s);
  logits[idx] = s;
}

// ------------------------ xp GEMM: C = A @ W^T + bias ----------------------
// A: [M, K] bf16 (optional strided row map), W: [768, K] bf16, C: [M, 768] bf16
// block = 256 threads (8 waves); block tile 128(M) x 64(N); wave = 16 x 64.

__global__ void __launch_bounds__(256)
gemm_xp_kernel(const bf16_t* __restrict__ A, const bf16_t* __restrict__ W,
               const float* __restrict__ bias, bf16_t* __restrict__ Cmat,
               int K, int a_row_stride, long a_row_off) {
  const int lane  = threadIdx.x & 31;
  const int wave  = threadIdx.x >> 5;
  const int mbase = blockIdx.x * 128 + wave * 16;
  const int nbase = blockIdx.y * 64;

  v8f acc[4];
  #pragma unroll
  for (int j = 0; j < 4; ++j)
    #pragma unroll
    for (int r = 0; r < 8; ++r) acc[j][r] = 0.0f;

  for (int k = 0; k < K; k += 32) {
    if (k + 32 < K)  // pull next W slice toward the WGP (j-loop reuses it)
      __builtin_prefetch((const void*)(W + (size_t)(nbase + (lane & 15)) * K
                                         + k + 32), 0, 3);
    const v16bf a = load_frag_a_rm(A, K, mbase, k, a_row_stride, a_row_off);
    #pragma unroll
    for (int j = 0; j < 4; ++j) {
      const v16bf b = load_frag_b(W, K, nbase + j * 16, k);
      acc[j] = wmma_bf16(a, b, acc[j]);
    }
  }

  #pragma unroll
  for (int j = 0; j < 4; ++j) {
    const int col = nbase + j * 16 + (lane & 15);
    const float bv = bias[col];
    #pragma unroll
    for (int r = 0; r < 8; ++r) {
      const int m = mbase + r + ((lane >> 4) << 3);
      Cmat[(size_t)m * kG + col] = (bf16_t)(acc[j][r] + bv);
    }
  }
}

// ----------------------------- recurrent scan ------------------------------
// One 1024-thread workgroup (32 wave32) per direction. h kept bf16 in LDS.
// Phase 1 works in "strips": 96 strips = 48 N-tiles x 2 M-halves; each wave
// owns 3 strips (= 12 tiles). Within a strip the B (W_hh) fragment is loaded
// once per k-chunk and reused across 4 WMMAs -> 4x less L2 traffic per step.

__global__ void __launch_bounds__(1024)
gru_scan_kernel(const bf16_t* __restrict__ xp_base, unsigned long long xp_dir_stride,
                long sb0, long st0, long sb1, long st1,
                const bf16_t* __restrict__ Whh_base,
                const float* __restrict__ bhh_base,
                bf16_t* __restrict__ out_seq,     // [B,S,2H] bf16 or null
                float* __restrict__ out_final,    // [B,2H] fp32 or null
                int steps_fwd, int steps_bwd) {
  extern __shared__ __align__(16) char smem[];
  bf16_t* hsh = (bf16_t*)smem;                       // [128][H_LD]
  bf16_t* gr  = (bf16_t*)(smem + kB * H_LD * 2);     // [128][256] sigmoid(r)
  bf16_t* gz  = gr + kB * kH;                        // [128][256] sigmoid(z)
  bf16_t* gn  = gz + kB * kH;                        // [128][256] hn (raw)

  const int dir = blockIdx.x;
  const bf16_t* xp  = xp_base + (size_t)dir * xp_dir_stride;
  const long sb = dir ? sb1 : sb0;
  const long st = dir ? st1 : st0;
  const bf16_t* Whh = Whh_base + (size_t)dir * kG * kH;
  const float*  bhh = bhh_base + dir * kG;
  const int steps   = dir ? steps_bwd : steps_fwd;

  const int tid  = threadIdx.x;
  const int lane = tid & 31;
  const int wave = tid >> 5;

  for (int i = tid; i < kB * H_LD; i += 1024) hsh[i] = (bf16_t)0.0f;
  __syncthreads();

  for (int kstep = 0; kstep < steps; ++kstep) {
    const int t = (dir == 0) ? kstep : (kS - 1 - kstep);
    const long xp_t = (long)t * st;

    // ---- Phase 1: gh = h @ Whh^T, 3 strips (1 N-tile x 4 M-tiles) per wave
    for (int g = 0; g < 3; ++g) {
      const int strip = wave * 3 + g;        // 0..95
      const int ns    = strip >> 1;          // N-tile 0..47
      const int mh    = strip & 1;           // M-half: m-tiles mh*4 .. mh*4+3
      const int nbase = ns << 4;

      v8f acc[4];
      #pragma unroll
      for (int mi = 0; mi < 4; ++mi)
        #pragma unroll
        for (int r = 0; r < 8; ++r) acc[mi][r] = 0.0f;

      #pragma unroll
      for (int kc = 0; kc < kH; kc += 32) {
        const v16bf b = load_frag_b(Whh, kH, nbase, kc);   // 1 load, 4 uses
        #pragma unroll
        for (int mi = 0; mi < 4; ++mi) {
          const v16bf a = load_frag_a(hsh, H_LD, (mh * 4 + mi) << 4, kc);
          acc[mi] = wmma_bf16(a, b, acc[mi]);
        }
      }

      const int col  = nbase + (lane & 15);
      const int gate = nbase >> 8;          // 0=r, 1=z, 2=n (wave-uniform)
      const int colH = col & 255;
      const float bv = bhh[col];
      #pragma unroll
      for (int mi = 0; mi < 4; ++mi) {
        const int mbase = (mh * 4 + mi) << 4;
        #pragma unroll
        for (int r = 0; r < 8; ++r) {
          const int bi = mbase + r + ((lane >> 4) << 3);   // batch index
          const float hval = acc[mi][r] + bv;
          if (gate == 2) {
            gn[bi * kH + colH] = (bf16_t)hval;             // tanh deferred
          } else {
            const float xv = (float)xp[(size_t)bi * sb + xp_t + col];
            const float s  = fast_sigmoid(xv + hval);
            if (gate == 0) gr[bi * kH + colH] = (bf16_t)s;
            else           gz[bi * kH + colH] = (bf16_t)s;
          }
        }
      }
    }
    __syncthreads();

    // ---- Phase 2: pointwise gate combine + h update -----------------------
    #pragma unroll
    for (int i = 0; i < (kB * kH) / 1024; ++i) {
      const int idx = tid + i * 1024;
      const int bi  = idx >> 8;
      const int c   = idx & 255;
      const float r  = (float)gr[idx];
      const float z  = (float)gz[idx];
      const float hn = (float)gn[idx];
      const float xn = (float)xp[(size_t)bi * sb + xp_t + 2 * kH + c];
      const float hp = (float)hsh[bi * H_LD + c];
      const float nv = fast_tanh(xn + r * hn);
      const float hnew = (1.0f - z) * nv + z * hp;
      hsh[bi * H_LD + c] = (bf16_t)hnew;
      if (out_seq)
        out_seq[((size_t)bi * kS + t) * (2 * kH) + dir * kH + c] = (bf16_t)hnew;
      if (out_final && kstep == steps - 1)
        out_final[bi * (2 * kH) + dir * kH + c] = hnew;
    }
    __syncthreads();
  }
}

// ------------------------------- launcher ----------------------------------

extern "C" void kernel_launch(void* const* d_in, const int* in_sizes, int n_in,
                              void* d_out, int out_size, void* d_ws, size_t ws_size,
                              hipStream_t stream) {
  (void)in_sizes; (void)n_in; (void)out_size; (void)ws_size;

  const float* coordinate = (const float*)d_in[0];
  const int*   flag_bits  = (const int*)d_in[1];
  const int*   pos_enc    = (const int*)d_in[2];
  const float* coord_W    = (const float*)d_in[3];
  const float* feat_emb   = (const float*)d_in[4];
  const float* W_ih_l0    = (const float*)d_in[5];
  const float* W_hh_l0    = (const float*)d_in[6];
  const float* b_ih_l0    = (const float*)d_in[7];
  const float* b_hh_l0    = (const float*)d_in[8];
  const float* W_ih_l1    = (const float*)d_in[9];
  const float* W_hh_l1    = (const float*)d_in[10];
  const float* b_ih_l1    = (const float*)d_in[11];
  const float* b_hh_l1    = (const float*)d_in[12];
  const float* out_W      = (const float*)d_in[13];
  const float* out_b      = (const float*)d_in[14];

  float* outp   = (float*)d_out;
  float* logits = outp;               // [128, 345]
  float* featur = outp + kB * kC;     // [128, 512]

  char* ws = (char*)d_ws;
  const size_t OFF_X  = 0;                                       // x bf16
  const size_t OFF_H1 = OFF_X  + (size_t)kB * kS * kE * 2;       // h1 bf16
  const size_t OFF_XP = OFF_H1 + (size_t)kB * kS * (2 * kH) * 2; // xp bf16 x2
  const size_t OFF_W  = OFF_XP + (size_t)2 * kB * kS * kG * 2;   // weights
  bf16_t* x_bf    = (bf16_t*)(ws + OFF_X);
  bf16_t* h1_bf   = (bf16_t*)(ws + OFF_H1);
  bf16_t* xp_bf   = (bf16_t*)(ws + OFF_XP);
  bf16_t* wih0_bf = (bf16_t*)(ws + OFF_W);
  bf16_t* whh0_bf = wih0_bf + (size_t)2 * kG * kE;
  bf16_t* wih1_bf = whh0_bf + (size_t)2 * kG * kH;
  bf16_t* whh1_bf = wih1_bf + (size_t)2 * kG * (2 * kH);

  hipFuncSetAttribute((const void*)gru_scan_kernel,
                      hipFuncAttributeMaxDynamicSharedMemorySize, SCAN_SMEM);

  // bf16 weight copies (deterministic; recomputed each call)
  convert_f32_bf16<<<512, 256, 0, stream>>>(W_ih_l0, wih0_bf, 2 * kG * kE);
  convert_f32_bf16<<<512, 256, 0, stream>>>(W_hh_l0, whh0_bf, 2 * kG * kH);
  convert_f32_bf16<<<1024, 256, 0, stream>>>(W_ih_l1, wih1_bf, 2 * kG * 2 * kH);
  convert_f32_bf16<<<512, 256, 0, stream>>>(W_hh_l1, whh1_bf, 2 * kG * kH);

  // embedding -> x (bf16)
  embed_kernel<<<kB * kS, kE, 0, stream>>>(coordinate, flag_bits, pos_enc,
                                           coord_W, feat_emb, x_bf);

  // layer 0 input-gate GEMMs: xp = x @ W_ih^T + b_ih   (both directions)
  dim3 gbig((kB * kS) / 128, kG / 64);
  gemm_xp_kernel<<<gbig, 256, 0, stream>>>(x_bf, wih0_bf, b_ih_l0,
                                           xp_bf, kE, 1, 0);
  gemm_xp_kernel<<<gbig, 256, 0, stream>>>(x_bf, wih0_bf + (size_t)kG * kE,
                                           b_ih_l0 + kG,
                                           xp_bf + (size_t)kB * kS * kG, kE, 1, 0);

  // layer 0 scan: fwd + bwd (full sequence output to h1)
  gru_scan_kernel<<<2, 1024, SCAN_SMEM, stream>>>(
      xp_bf, (unsigned long long)kB * kS * kG,
      (long)kS * kG, (long)kG, (long)kS * kG, (long)kG,
      whh0_bf, b_hh_l0, h1_bf, nullptr, kS, kS);

  // layer 1 input-gate GEMMs.  Bwd direction only needs original t = S-1
  // (its output at position -1 is the first scan step), so M=128 rows.
  gemm_xp_kernel<<<gbig, 256, 0, stream>>>(h1_bf, wih1_bf, b_ih_l1,
                                           xp_bf, 2 * kH, 1, 0);
  dim3 gsmall(1, kG / 64);
  gemm_xp_kernel<<<gsmall, 256, 0, stream>>>(h1_bf,
                                             wih1_bf + (size_t)kG * (2 * kH),
                                             b_ih_l1 + kG,
                                             xp_bf + (size_t)kB * kS * kG,
                                             2 * kH, kS, kS - 1);

  // layer 1 scan: fwd full 512 steps, bwd exactly 1 step; only final h kept.
  gru_scan_kernel<<<2, 1024, SCAN_SMEM, stream>>>(
      xp_bf, (unsigned long long)kB * kS * kG,
      (long)kS * kG, (long)kG, (long)kG, 0L,
      whh1_bf, b_hh_l1, nullptr, featur, kS, 1);

  // logits = featur @ out_W^T + out_b
  logits_kernel<<<(kB * kC + 255) / 256, 256, 0, stream>>>(featur, out_W,
                                                           out_b, logits);
}
